// Phase3SatGate_71356586655922
// MI455X (gfx1250) — matrix operand order repaired and checked by reading, not compiled, not run on hardware
//
#include <hip/hip_runtime.h>

// ---------------------------------------------------------------------------
// MI455X (gfx1250) ConvLSTM pipeline.
// Dominant cost: 12x conv3x3 (Cin=24 -> Cout=64) over 8x256x256  (~174 GFLOP).
// Implemented as implicit GEMM on V_WMMA_F32_16X16X32_F16:
//   M = 64 out-channels  (4 tiles == the 4 LSTM gates i/f/o/g)
//   N = pixels           (16x16 tile per workgroup, 16-pixel rows per WMMA)
//   K = 24 in-ch per tap (padded to 32; 9 accumulating WMMAs = 3x3 taps)
// Activations/weights f16, accumulation f32. LSTM nonlinearity fully in
// registers (gate tiles are lane/VGPR aligned). h ping-pongs (halo reads).
// ---------------------------------------------------------------------------

typedef __attribute__((ext_vector_type(16))) _Float16 v16h;
typedef __attribute__((ext_vector_type(8)))  float    v8f;

#define BB 8
#define TT 12
#define HH 256
#define WW 256
#define ECH 8
#define HCH 16
#define NPIX (HH*WW)

__device__ __forceinline__ float sig_f(float x) { return 1.0f / (1.0f + __expf(-x)); }
__device__ __forceinline__ float tanh_f(float x) { return 2.0f / (1.0f + __expf(-2.0f*x)) - 1.0f; }

// ---------------------------------------------------------------------------
// Zero h0 (f16) and c (f32) state: n = B*H*W*HC elements each.
// ---------------------------------------------------------------------------
__global__ __launch_bounds__(256) void zero_state_kernel(float* __restrict__ c,
                                                         _Float16* __restrict__ h0,
                                                         int n) {
  int i = blockIdx.x * 256 + threadIdx.x;
  if (i < n) { c[i] = 0.0f; h0[i] = (_Float16)0.0f; }
}

// ---------------------------------------------------------------------------
// Pack cell_w (64,24,3,3) f32 into WMMA A-matrix per-lane layout, f16.
// pw index: ((tap*4 + m)*32 + lane)*16 + j
//   A layout (contiguous K): halves 0..15 <- K = (lane<16 ? 0..15 : 16..31)
//   M = m*16 + lane%16 ; K = in-channel (0..23 valid, 24..31 zero pad)
// ---------------------------------------------------------------------------
__global__ __launch_bounds__(256) void pack_w_kernel(const float* __restrict__ cw,
                                                     _Float16* __restrict__ pw) {
  int i = blockIdx.x * 256 + threadIdx.x;
  if (i >= 9*4*32*16) return;
  int j    = i & 15;
  int lane = (i >> 4) & 31;
  int m    = (i >> 9) & 3;
  int tap  = i >> 11;
  int hi   = lane >> 4;
  int M    = m*16 + (lane & 15);
  int k    = hi*16 + j;
  float v  = (k < 24) ? cw[(M*24 + k)*9 + tap] : 0.0f;
  pw[i] = (_Float16)v;
}

// ---------------------------------------------------------------------------
// SAT scalar gate: alpha[b] = prior(month) * sigmoid(MLP(sat)).
// ---------------------------------------------------------------------------
__global__ void gate_kernel(const float* __restrict__ sat, const int* __restrict__ months,
                            const float* __restrict__ w1, const float* __restrict__ b1,
                            const float* __restrict__ w2, const float* __restrict__ b2,
                            float* __restrict__ alpha) {
  int b = threadIdx.x;
  if (b >= BB) return;
  float s = sat[b];
  float acc = b2[0];
  #pragma unroll
  for (int j = 0; j < 8; ++j) acc += w2[j] * fmaxf(s * w1[j] + b1[j], 0.0f);
  float a = sig_f(acc);
  int mo = months[b];
  float prior = (mo >= 3 && mo <= 6) ? 0.6f : 0.3f;
  alpha[b] = prior * a;
}

// ---------------------------------------------------------------------------
// Encoder: relu(conv2(relu(conv1(x)))) per frame, fused through LDS.
// Output enc: (B*T, H, W, EC) f16 (channel-last for WMMA-side consumption).
// grid = (256 tiles, 96 frames), block = 256.
// ---------------------------------------------------------------------------
__global__ __launch_bounds__(256)
void encoder_kernel(const float* __restrict__ x,
                    const float* __restrict__ w1, const float* __restrict__ b1,
                    const float* __restrict__ w2, const float* __restrict__ b2,
                    _Float16* __restrict__ enc) {
  __shared__ float xt[20*20];
  __shared__ float t1[18*18*ECH];
  __shared__ float w1s[72], b1s[8], w2s[576], b2s[8];

  const int tid  = threadIdx.x;
  const int f    = blockIdx.y;                 // frame = b*T + t
  const int tile = blockIdx.x;
  const int ty0  = (tile >> 4) << 4;
  const int tx0  = (tile & 15) << 4;

  if (tid < 72) w1s[tid] = w1[tid];
  if (tid < 8) { b1s[tid] = b1[tid]; b2s[tid] = b2[tid]; }
  for (int i = tid; i < 576; i += 256) w2s[i] = w2[i];

  const float* xf = x + (size_t)f * NPIX;
  for (int p = tid; p < 400; p += 256) {
    int ys = p / 20, xs = p % 20;
    int gy = ty0 - 2 + ys, gx = tx0 - 2 + xs;
    xt[p] = (gy >= 0 && gy < HH && gx >= 0 && gx < WW) ? xf[gy*WW + gx] : 0.0f;
  }
  __syncthreads();

  for (int q = tid; q < 324; q += 256) {
    int ys = q / 18, xs = q % 18;
    #pragma unroll
    for (int oc = 0; oc < ECH; ++oc) {
      float s = b1s[oc];
      #pragma unroll
      for (int ky = 0; ky < 3; ++ky)
        #pragma unroll
        for (int kx = 0; kx < 3; ++kx)
          s += w1s[oc*9 + ky*3 + kx] * xt[(ys+ky)*20 + xs + kx];
      t1[q*ECH + oc] = fmaxf(s, 0.0f);
    }
  }
  __syncthreads();

  const int oy = tid >> 4, ox = tid & 15;
  __align__(16) _Float16 ov[8];
  #pragma unroll
  for (int oc = 0; oc < ECH; ++oc) {
    float s = b2s[oc];
    #pragma unroll
    for (int ky = 0; ky < 3; ++ky)
      #pragma unroll
      for (int kx = 0; kx < 3; ++kx) {
        const float* tp = &t1[((oy+ky)*18 + ox + kx)*ECH];
        const float* wp = &w2s[(oc*ECH)*9 + ky*3 + kx];
        #pragma unroll
        for (int ic = 0; ic < ECH; ++ic) s += wp[ic*9] * tp[ic];
      }
    ov[oc] = (_Float16)fmaxf(s, 0.0f);
  }
  size_t idx = ((((size_t)f * HH) + (ty0 + oy)) * WW + (tx0 + ox)) * ECH;
  *(uint4*)(enc + idx) = *(const uint4*)ov;
}

// ---------------------------------------------------------------------------
// ConvLSTM step (the hot kernel). grid = (256 tiles, B), block = 256 (8 waves).
// enc_t(8ch f16) + h_in(16ch f16) staged channel-last into LDS with 1-pixel
// halo, channels padded to 32 (zeros). 72 wmma per wave; LSTM in registers.
// ---------------------------------------------------------------------------
__global__ __launch_bounds__(256)
void cell_kernel(const _Float16* __restrict__ enc,    // (B,T,H,W,EC) f16
                 const _Float16* __restrict__ h_in,   // (B,H,W,HC)   f16
                 float* __restrict__ c_st,            // (B,H,W,HC)   f32 (in/out)
                 _Float16* __restrict__ h_out,        // (B,H,W,HC)   f16
                 const _Float16* __restrict__ pw,     // packed A weights
                 const float* __restrict__ cell_b,
                 int t) {
  __shared__ __attribute__((aligned(32))) _Float16 in_t[18*18*32];   // 20736 B
  __shared__ __attribute__((aligned(32))) _Float16 w_l[9*4*32*16];   // 36864 B

  const int tid  = threadIdx.x;
  const int b    = blockIdx.y;
  const int tile = blockIdx.x;
  const int ty0  = (tile >> 4) << 4;
  const int tx0  = (tile & 15) << 4;

  // Stage packed weights (L2-resident, 36 KB).
  {
    const uint4* src = (const uint4*)pw;
    uint4* dst = (uint4*)w_l;
    #pragma unroll
    for (int i = 0; i < 9; ++i) dst[tid + i*256] = src[tid + i*256];
  }

  // Stage input tile + halo: [pix 18x18][ch: 8 enc | 16 h | 8 zero] f16.
  const _Float16* encb = enc + ((size_t)(b*TT + t)) * NPIX * ECH;
  const _Float16* hb   = h_in + (size_t)b * NPIX * HCH;
  for (int p = tid; p < 18*18; p += 256) {
    int ys = p / 18, xs = p % 18;
    int gy = ty0 - 1 + ys, gx = tx0 - 1 + xs;
    uint4 e0 = {0,0,0,0}, h0 = {0,0,0,0}, h1 = {0,0,0,0}, zz = {0,0,0,0};
    if (gy >= 0 && gy < HH && gx >= 0 && gx < WW) {
      int pi = gy*WW + gx;
      e0 = *(const uint4*)(encb + (size_t)pi*ECH);
      const uint4* hp = (const uint4*)(hb + (size_t)pi*HCH);
      h0 = hp[0]; h1 = hp[1];
    }
    uint4* d = (uint4*)(in_t + p*32);
    d[0] = e0; d[1] = h0; d[2] = h1; d[3] = zz;
  }
  __syncthreads();

  const int lane  = tid & 31;
  const int wv    = tid >> 5;          // 0..7
  const int y0    = wv * 2;            // this wave owns rows y0, y0+1
  const int xl    = lane & 15;         // N = pixel x within row
  const int khalf = lane >> 4;         // B: K half (0-15 | 16-31)

  const v16h* inv = (const v16h*)in_t; // element i = halves [16i,16i+16)
  const v16h* wl  = (const v16h*)w_l;

  v8f acc[4][2];
  {
    v8f z = {};
    #pragma unroll
    for (int m = 0; m < 4; ++m) { acc[m][0] = z; acc[m][1] = z; }
  }

  #pragma unroll
  for (int ky = 0; ky < 3; ++ky) {
    #pragma unroll
    for (int kx = 0; kx < 3; ++kx) {
      const int tap = ky*3 + kx;
      v16h b0 = inv[((y0     + ky)*18 + xl + kx)*2 + khalf];
      v16h b1 = inv[((y0 + 1 + ky)*18 + xl + kx)*2 + khalf];
      #pragma unroll
      for (int m = 0; m < 4; ++m) {
        v16h av = wl[(tap*4 + m)*32 + lane];
        acc[m][0] = __builtin_amdgcn_wmma_f32_16x16x32_f16(
            false, av, false, b0, (short)0, acc[m][0], false, false);
        acc[m][1] = __builtin_amdgcn_wmma_f32_16x16x32_f16(
            false, av, false, b1, (short)0, acc[m][1], false, false);
      }
    }
  }

  // Epilogue: gate bias + LSTM nonlinearity, entirely in registers.
  // D layout: VGPR e, lane L -> M = e + 8*(L/16) (= channel within gate),
  //           N = L%16 (= x). Lane covers 8 contiguous channels cb..cb+7.
  const int cb = khalf * 8;
  float bi[4][8];
  #pragma unroll
  for (int g = 0; g < 4; ++g)
    #pragma unroll
    for (int e = 0; e < 8; ++e) bi[g][e] = cell_b[g*16 + cb + e];

  #pragma unroll
  for (int r = 0; r < 2; ++r) {
    const int gy = ty0 + y0 + r;
    const int gx = tx0 + xl;
    const size_t base = (((size_t)b*HH + gy)*WW + gx)*HCH + cb;

    __align__(16) float cv[8];
    { const float4* cp = (const float4*)(c_st + base);
      *(float4*)&cv[0] = cp[0]; *(float4*)&cv[4] = cp[1]; }

    __align__(16) _Float16 hv[8];
    #pragma unroll
    for (int e = 0; e < 8; ++e) {
      float ig = sig_f(acc[0][r][e] + bi[0][e]);
      float fg = sig_f(acc[1][r][e] + bi[1][e]);
      float og = sig_f(acc[2][r][e] + bi[2][e]);
      float gg = tanh_f(acc[3][r][e] + bi[3][e]);
      float cn = fg * cv[e] + ig * gg;
      cv[e] = cn;
      hv[e] = (_Float16)(og * tanh_f(cn));
    }
    { float4* cp = (float4*)(c_st + base);
      cp[0] = *(const float4*)&cv[0]; cp[1] = *(const float4*)&cv[4]; }
    *(uint4*)(h_out + base) = *(const uint4*)hv;
  }
}

// ---------------------------------------------------------------------------
// Aux winter branch: h_aux = wp_w @ mean_t(enc) + wp_b  (f16 out, NHWC).
// ---------------------------------------------------------------------------
__global__ __launch_bounds__(256)
void aux_kernel(const _Float16* __restrict__ enc,
                const float* __restrict__ wp_w, const float* __restrict__ wp_b,
                _Float16* __restrict__ haux) {
  int p = blockIdx.x * 256 + threadIdx.x;        // 0 .. B*NPIX-1
  int b = p >> 16;
  int pix = p & (NPIX - 1);
  float acc[ECH] = {};
  for (int t = 0; t < TT; ++t) {
    const _Float16* e = enc + (((size_t)(b*TT + t)) * NPIX + pix) * ECH;
    #pragma unroll
    for (int ec = 0; ec < ECH; ++ec) acc[ec] += (float)e[ec];
  }
  const float inv_t = 1.0f / (float)TT;
  __align__(16) _Float16 o[HCH];
  #pragma unroll
  for (int hc = 0; hc < HCH; ++hc) {
    float s = wp_b[hc];
    #pragma unroll
    for (int ec = 0; ec < ECH; ++ec) s += wp_w[hc*ECH + ec] * (acc[ec] * inv_t);
    o[hc] = (_Float16)s;
  }
  uint4* d = (uint4*)(haux + (size_t)p * HCH);
  d[0] = ((const uint4*)o)[0];
  d[1] = ((const uint4*)o)[1];
}

// ---------------------------------------------------------------------------
// Head: out = sigmoid(head_w . (h_main + alpha*h_aux) + head_b), (B,1,H,W) f32.
// ---------------------------------------------------------------------------
__global__ __launch_bounds__(256)
void head_kernel(const _Float16* __restrict__ h, const _Float16* __restrict__ haux,
                 const float* __restrict__ alpha,
                 const float* __restrict__ head_w, const float* __restrict__ head_b,
                 float* __restrict__ out) {
  int p = blockIdx.x * 256 + threadIdx.x;
  int b = p >> 16;
  float al = alpha[b];
  const _Float16* hp = h + (size_t)p * HCH;
  const _Float16* ap = haux + (size_t)p * HCH;
  float s = head_b[0];
  #pragma unroll
  for (int hc = 0; hc < HCH; ++hc)
    s += head_w[hc] * ((float)hp[hc] + al * (float)ap[hc]);
  out[p] = sig_f(s);
}

// ---------------------------------------------------------------------------
// Launch.  Workspace layout (~185 MB):
//   enc (B,T,H,W,EC) f16 | packed W f16 | h ping | h pong | c f32 | h_aux f16 | alpha
// ---------------------------------------------------------------------------
extern "C" void kernel_launch(void* const* d_in, const int* in_sizes, int n_in,
                              void* d_out, int out_size, void* d_ws, size_t ws_size,
                              hipStream_t stream) {
  (void)in_sizes; (void)n_in; (void)out_size; (void)ws_size;

  const float* x       = (const float*)d_in[0];
  const float* sat     = (const float*)d_in[1];
  const int*   months  = (const int*)  d_in[2];
  const float* enc_w1  = (const float*)d_in[3];
  const float* enc_b1  = (const float*)d_in[4];
  const float* enc_w2  = (const float*)d_in[5];
  const float* enc_b2  = (const float*)d_in[6];
  const float* cell_w  = (const float*)d_in[7];
  const float* cell_b  = (const float*)d_in[8];
  const float* wp_w    = (const float*)d_in[9];
  const float* wp_b    = (const float*)d_in[10];
  const float* head_w  = (const float*)d_in[11];
  const float* head_b  = (const float*)d_in[12];
  const float* sg_w1   = (const float*)d_in[13];
  const float* sg_b1   = (const float*)d_in[14];
  const float* sg_w2   = (const float*)d_in[15];
  const float* sg_b2   = (const float*)d_in[16];
  float* out = (float*)d_out;

  char* ws = (char*)d_ws;
  size_t off = 0;
  auto take = [&](size_t bytes) -> char* {
    char* p = ws + off;
    off = (off + bytes + 255) & ~(size_t)255;
    return p;
  };
  const size_t n_state = (size_t)BB * NPIX * HCH;         // 8,388,608
  _Float16* enc   = (_Float16*)take((size_t)BB*TT*NPIX*ECH*sizeof(_Float16));
  _Float16* pw    = (_Float16*)take((size_t)9*4*32*16*sizeof(_Float16));
  _Float16* hbuf0 = (_Float16*)take(n_state*sizeof(_Float16));
  _Float16* hbuf1 = (_Float16*)take(n_state*sizeof(_Float16));
  float*    c_st  = (float*)   take(n_state*sizeof(float));
  _Float16* haux  = (_Float16*)take(n_state*sizeof(_Float16));
  float*    alpha = (float*)   take(32);
  _Float16* hbufs[2] = { hbuf0, hbuf1 };

  // 1) zero h0 + c
  zero_state_kernel<<<(int)((n_state + 255)/256), 256, 0, stream>>>(c_st, hbuf0, (int)n_state);
  // 2) pack cell weights into WMMA A layout
  pack_w_kernel<<<(9*4*32*16 + 255)/256, 256, 0, stream>>>(cell_w, pw);
  // 3) scalar SAT gate
  gate_kernel<<<1, 32, 0, stream>>>(sat, months, sg_w1, sg_b1, sg_w2, sg_b2, alpha);
  // 4) encoder over all B*T frames
  encoder_kernel<<<dim3(256, BB*TT), 256, 0, stream>>>(x, enc_w1, enc_b1, enc_w2, enc_b2, enc);
  // 5) aux branch (mean over T + 1x1 conv)
  aux_kernel<<<(BB*NPIX)/256, 256, 0, stream>>>(enc, wp_w, wp_b, haux);
  // 6) ConvLSTM recurrence (WMMA hot loop); h ping-pongs, final h lands in hbuf0
  for (int t = 0; t < TT; ++t) {
    cell_kernel<<<dim3(256, BB), 256, 0, stream>>>(
        enc, hbufs[t & 1], c_st, hbufs[(t + 1) & 1], pw, cell_b, t);
  }
  // 7) fuse + head + sigmoid
  head_kernel<<<(BB*NPIX)/256, 256, 0, stream>>>(hbufs[0], haux, alpha, head_w, head_b, out);
}
